// ChaosHydra_24395414241731
// MI455X (gfx1250) — compile-verified, matrix operand order checked by reading, not run on hardware
//
#include <hip/hip_runtime.h>

typedef __bf16 bf16;
typedef __attribute__((ext_vector_type(16))) __bf16 v16bf;
typedef __attribute__((ext_vector_type(8)))  __bf16 v8bf;
typedef __attribute__((ext_vector_type(8)))  float  v8f;

// Explicit global-address-space pointers so operand loads lower to
// global_load_b128 (LOADcnt only) instead of flat_load_b128 (LOADcnt+DScnt),
// even with loop-carried pointer increments.
typedef __attribute__((address_space(1))) const v8bf* gv8p;
typedef __attribute__((address_space(1))) float*      gf32p;
typedef __attribute__((address_space(1))) bf16*       gbf16p;

namespace {

constexpr int kB      = 2;
constexpr int kL      = 1024;
constexpr int kH      = 4;
constexpr int kDModel = 1024;
constexpr int kDInner = 2048;
constexpr int kM      = kB * kL;      // 2048 tokens
constexpr int kDtRank = 64;
constexpr int kDState = 16;
constexpr int kHid    = 512;
constexpr int kOut    = 1026;
constexpr int kOutPad = 1040;         // next multiple of 16

__device__ __forceinline__ bf16 f2bf(float f) {
  unsigned u = __builtin_bit_cast(unsigned, f);
  u += 0x7FFFu + ((u >> 16) & 1u);                 // round-to-nearest-even
  unsigned short h = (unsigned short)(u >> 16);
  return __builtin_bit_cast(bf16, h);
}

__device__ __forceinline__ v8f zero8() {
  v8f z;
#pragma unroll
  for (int i = 0; i < 8; ++i) z[i] = 0.f;
  return z;
}

__device__ __forceinline__ void stg(float* p, float v) { *(gf32p)p = v; }
__device__ __forceinline__ void stg(bf16* p, bf16 v)   { *(gbf16p)p = v; }

// Load one 16x32 bf16 WMMA operand row-slice at element offset `off` from a
// per-lane base pointer already offset by half*8.
// Layout per ISA 7.12.2: lanes 0-15 hold K {0..7,16..23}, lanes 16-31 hold
// K {8..15,24..31} of row (lane&15).
__device__ __forceinline__ v16bf ld_op(const bf16* __restrict__ p, int off) {
  const v8bf lo = *(gv8p)(p + off);
  const v8bf hi = *(gv8p)(p + off + 16);
  v16bf r;
#pragma unroll
  for (int i = 0; i < 8; ++i) { r[i] = lo[i]; r[i + 8] = hi[i]; }
  return r;
}

enum {
  MODE_F32 = 0,
  MODE_BF16 = 1,
  MODE_BIAS_LEAKY_BF16 = 2,
  MODE_BIAS_LEAKY_F32 = 3,
  MODE_SOFTPLUS = 4,
  MODE_LOGITS = 5
};

// C(MxN) = epilogue(A(MxK,bf16) * W(NxK,bf16)^T + bias)
// Wave: one 16-row M tile x 4 N tiles (A reused 4x). Block: 8 waves -> 128 rows.
// K-loop: branch-free, unroll-2 ping-pong double buffering (K/32 is even for
// every GEMM here), pointer-increment addressing with immediate load offsets.
// Out-of-range N tiles are clamped (compute discarded at store).
template <int MODE>
__global__ __launch_bounds__(256) void gemm_kernel(
    const bf16* __restrict__ A, int lda, long long strideA,
    const bf16* __restrict__ W, int ldw, long long strideW,
    const float* __restrict__ bias, int biasStride,
    void* __restrict__ C, int ldc, long long strideC,
    int M, int N, int K, int nOut) {
  const int lane = threadIdx.x & 31;
  const int wave = threadIdx.x >> 5;
  const int half = lane >> 4;
  const int lr   = lane & 15;
  const int z    = blockIdx.z;

  A += (long long)z * strideA;
  W += (long long)z * strideW;
  if (bias) bias += (long long)z * biasStride;

  const int mTile  = blockIdx.y * 8 + wave;
  const int nTile0 = blockIdx.x * 4;
  const int nTiles = (N + 15) >> 4;
  if (mTile * 16 >= M) return;

  // Per-lane operand pointers, pre-offset by half*8 (K sub-chunk select).
  const bf16* aPtr = A + (long long)(mTile * 16 + lr) * lda + half * 8;
  const bf16* wPtr[4];
#pragma unroll
  for (int j = 0; j < 4; ++j) {
    int nt = nTile0 + j;
    if (nt >= nTiles) nt = nTiles - 1;         // clamp: redundant compute, store-guarded
    wPtr[j] = W + (long long)(nt * 16 + lr) * ldw + half * 8;
  }

  v8f acc[4];
#pragma unroll
  for (int j = 0; j < 4; ++j) acc[j] = zero8();

  const int halfSteps = K >> 6;                // K/64; K/32 is even everywhere

  // Prologue: fill both buffers (steps 0 and 1).
  v16bf a0 = ld_op(aPtr, 0);
  v16bf a1 = ld_op(aPtr, 32);
  v16bf b0[4], b1[4];
#pragma unroll
  for (int j = 0; j < 4; ++j) { b0[j] = ld_op(wPtr[j], 0); b1[j] = ld_op(wPtr[j], 32); }
  aPtr += 64;
#pragma unroll
  for (int j = 0; j < 4; ++j) wPtr[j] += 64;

  for (int s = 0; s < halfSteps - 1; ++s) {
#pragma unroll
    for (int j = 0; j < 4; ++j)
      acc[j] = __builtin_amdgcn_wmma_f32_16x16x32_bf16(
          false, a0, false, b0[j], (short)0, acc[j], false, false);
    a0 = ld_op(aPtr, 0);
#pragma unroll
    for (int j = 0; j < 4; ++j) b0[j] = ld_op(wPtr[j], 0);
#pragma unroll
    for (int j = 0; j < 4; ++j)
      acc[j] = __builtin_amdgcn_wmma_f32_16x16x32_bf16(
          false, a1, false, b1[j], (short)0, acc[j], false, false);
    a1 = ld_op(aPtr, 32);
#pragma unroll
    for (int j = 0; j < 4; ++j) b1[j] = ld_op(wPtr[j], 32);
    aPtr += 64;
#pragma unroll
    for (int j = 0; j < 4; ++j) wPtr[j] += 64;
  }
  // Epilogue computes (last two steps).
#pragma unroll
  for (int j = 0; j < 4; ++j)
    acc[j] = __builtin_amdgcn_wmma_f32_16x16x32_bf16(
        false, a0, false, b0[j], (short)0, acc[j], false, false);
#pragma unroll
  for (int j = 0; j < 4; ++j)
    acc[j] = __builtin_amdgcn_wmma_f32_16x16x32_bf16(
        false, a1, false, b1[j], (short)0, acc[j], false, false);

  float* Cf = (float*)C;
  bf16*  Cb = (bf16*)C;
#pragma unroll
  for (int j = 0; j < 4; ++j) {
    if (nTile0 + j >= nTiles) continue;        // store guard only (not in K loop)
    const int nCol = (nTile0 + j) * 16 + lr;
    float bv = 0.f;
    if (MODE == MODE_BIAS_LEAKY_BF16 || MODE == MODE_BIAS_LEAKY_F32 ||
        MODE == MODE_SOFTPLUS || MODE == MODE_LOGITS) {
      if (nCol < nOut) bv = bias[nCol];
    }
#pragma unroll
    for (int i = 0; i < 8; ++i) {
      const int r = mTile * 16 + half * 8 + i; // D-layout: VGPR i -> M=i+8*half
      float v = acc[j][i];
      if (MODE == MODE_F32) {
        stg(Cf + (long long)z * strideC + (long long)r * ldc + nCol, v);
      } else if (MODE == MODE_BF16) {
        stg(Cb + (long long)z * strideC + (long long)r * ldc + nCol, f2bf(v));
      } else if (MODE == MODE_BIAS_LEAKY_BF16) {
        v += bv; v = (v >= 0.f) ? v : 0.01f * v;
        stg(Cb + (long long)z * strideC + (long long)r * ldc + nCol, f2bf(v));
      } else if (MODE == MODE_BIAS_LEAKY_F32) {
        v += bv; v = (v >= 0.f) ? v : 0.01f * v;
        stg(Cf + (long long)z * strideC + (long long)r * ldc + nCol, v);
      } else if (MODE == MODE_SOFTPLUS) {
        v += bv; v = (v > 20.f) ? v : log1pf(__expf(v));
        stg(Cf + (long long)z * strideC + (long long)r * ldc + nCol, v);
      } else if (MODE == MODE_LOGITS) {
        if (nCol < nOut) {
          v += bv;
          const int bq = r >> 10, l = r & 1023;  // m = b*L + l
          // pred_logits[b, n, l, h]
          stg(Cf + (((long long)bq * kOut + nCol) * kL + l) * kH + z, v);
        }
      }
    }
  }
}

__global__ void embed_kernel(const int* __restrict__ ids, const float* __restrict__ emb,
                             bf16* __restrict__ u) {
  const int m = blockIdx.x;
  for (int idx = threadIdx.x; idx < kDModel; idx += blockDim.x) {
    const int h = idx >> 8, c = idx & 255;
    const int id = ids[m * kH + h];
    u[(long long)m * kDModel + idx] = f2bf(emb[((long long)h * kOut + id) * 256 + c]);
  }
}

__global__ void cvt_bf16_kernel(const float* __restrict__ src, bf16* __restrict__ dst,
                                long long n) {
  const long long gid = (long long)blockIdx.x * blockDim.x + threadIdx.x;
  if (gid < n) dst[gid] = f2bf(src[gid]);
}

__global__ void cvt_strided_kernel(const float* __restrict__ src, int ldSrc,
                                   bf16* __restrict__ dst, int ldDst, int rows, int cols) {
  const long long gid = (long long)blockIdx.x * blockDim.x + threadIdx.x;
  if (gid >= (long long)rows * cols) return;
  const int r = (int)(gid / cols), c = (int)(gid % cols);
  dst[(long long)r * ldDst + c] = f2bf(src[(long long)r * ldSrc + c]);
}

// (K,N) f32 per head -> (Npad,K) bf16 per head; pad rows zeroed.
__global__ void transpose_w_kernel(const float* __restrict__ src, bf16* __restrict__ dst,
                                   int K, int N, int Npad) {
  const int z = blockIdx.z;
  src += (long long)z * K * N;
  dst += (long long)z * Npad * K;
  const long long gid = (long long)blockIdx.x * blockDim.x + threadIdx.x;
  if (gid >= (long long)Npad * K) return;
  const int n = (int)(gid / K), k = (int)(gid % K);
  dst[gid] = (n < N) ? f2bf(src[(long long)k * N + n]) : f2bf(0.f);
}

// causal depthwise conv(4) + SiLU; writes f32 and bf16 copies
__global__ void conv_silu_kernel(const float* __restrict__ xz, const float* __restrict__ cw,
                                 const float* __restrict__ cb, float* __restrict__ xs,
                                 bf16* __restrict__ xsb) {
  const long long gid = (long long)blockIdx.x * blockDim.x + threadIdx.x;
  if (gid >= (long long)kM * kDInner) return;
  const int d = (int)(gid & (kDInner - 1));
  const int m = (int)(gid >> 11);
  const int l = m & (kL - 1);
  float acc = cb[d];
#pragma unroll
  for (int j = 0; j < 4; ++j) {
    const int lj = l - 3 + j;
    if (lj >= 0) acc = fmaf(cw[d * 4 + j], xz[(long long)(m - 3 + j) * 4096 + d], acc);
  }
  const float s = acc / (1.f + __expf(-acc));
  xs[gid] = s;
  xsb[gid] = f2bf(s);
}

// Selective scan: one (b,d) channel per half-wave, one state per lane.
// h_s(t) = exp(dt*A_s)*h_s + dt*x*B_s ; y = sum_s h_s*C_s via shfl_xor tree.
__global__ __launch_bounds__(256) void scan_kernel(
    const float* __restrict__ dt, const float* __restrict__ xs,
    const float* __restrict__ dbl, const float* __restrict__ A_log,
    float* __restrict__ y) {
  const int lane = threadIdx.x & 31;
  const int wave = threadIdx.x >> 5;
  const int half = lane >> 4;
  const int s    = lane & 15;
  const int ch = blockIdx.x * 16 + wave * 2 + half;   // 0..4095
  const int b = ch >> 11, d = ch & (kDInner - 1);
  const float a = -__expf(A_log[d * kDState + s]);
  float h = 0.f;
  const int base = b * kL;
  for (int t = 0; t < kL; ++t) {
    const int m = base + t;
    const float dtv = dt[(long long)m * kDInner + d];
    const float xv  = xs[(long long)m * kDInner + d];
    const float Bv  = dbl[m * 96 + kDtRank + s];
    const float Cv  = dbl[m * 96 + kDtRank + kDState + s];
    h = __expf(dtv * a) * h + dtv * xv * Bv;
    float p = h * Cv;
    p += __shfl_xor(p, 1); p += __shfl_xor(p, 2);
    p += __shfl_xor(p, 4); p += __shfl_xor(p, 8);
    if (s == 0) y[(long long)m * kDInner + d] = p;
  }
}

// y2 = (y + xs*D) * silu(z) -> bf16
__global__ void ygate_kernel(const float* __restrict__ y, const float* __restrict__ xs,
                             const float* __restrict__ Dp, const float* __restrict__ xz,
                             bf16* __restrict__ y2) {
  const long long gid = (long long)blockIdx.x * blockDim.x + threadIdx.x;
  if (gid >= (long long)kM * kDInner) return;
  const int d = (int)(gid & (kDInner - 1));
  const int m = (int)(gid >> 11);
  const float v = y[gid] + xs[gid] * Dp[d];
  const float zz = xz[(long long)m * 4096 + kDInner + d];
  y2[gid] = f2bf(v * zz / (1.f + __expf(-zz)));
}

__global__ __launch_bounds__(256) void layernorm_kernel(
    const float* __restrict__ X, long long strideX,
    const float* __restrict__ g, const float* __restrict__ b, int strideP,
    bf16* __restrict__ Y, long long strideY, int N) {
  __shared__ float red[2][8];
  const int row = blockIdx.x;
  const int z   = blockIdx.y;
  const float* x = X + (long long)z * strideX + (long long)row * N;
  float s = 0.f, s2 = 0.f;
  for (int i = threadIdx.x; i < N; i += 256) {
    const float v = x[i];
    s += v; s2 += v * v;
  }
#pragma unroll
  for (int m = 16; m; m >>= 1) { s += __shfl_xor(s, m); s2 += __shfl_xor(s2, m); }
  const int wave = threadIdx.x >> 5, lane = threadIdx.x & 31;
  if (lane == 0) { red[0][wave] = s; red[1][wave] = s2; }
  __syncthreads();
  if (wave == 0) {
    s  = (lane < 8) ? red[0][lane] : 0.f;
    s2 = (lane < 8) ? red[1][lane] : 0.f;
#pragma unroll
    for (int m = 4; m; m >>= 1) { s += __shfl_xor(s, m); s2 += __shfl_xor(s2, m); }
    if (lane == 0) { red[0][0] = s; red[1][0] = s2; }
  }
  __syncthreads();
  const float mean = red[0][0] / N;
  const float var  = red[1][0] / N - mean * mean;
  const float inv  = rsqrtf(var + 1e-5f);
  const float* gg = g + (long long)z * strideP;
  const float* bb = b + (long long)z * strideP;
  bf16* yo = Y + (long long)z * strideY + (long long)row * N;
  for (int i = threadIdx.x; i < N; i += 256)
    yo[i] = f2bf((x[i] - mean) * inv * gg[i] + bb[i]);
}

}  // namespace

extern "C" void kernel_launch(void* const* d_in, const int* in_sizes, int n_in,
                              void* d_out, int out_size, void* d_ws, size_t ws_size,
                              hipStream_t stream) {
  (void)in_sizes; (void)n_in; (void)out_size; (void)ws_size;
  const int*   x_ids    = (const int*)  d_in[0];
  const float* emb      = (const float*)d_in[1];
  const float* in_proj  = (const float*)d_in[2];
  const float* conv_w   = (const float*)d_in[3];
  const float* conv_b   = (const float*)d_in[4];
  const float* x_proj   = (const float*)d_in[5];
  const float* dt_proj  = (const float*)d_in[6];
  const float* dt_b     = (const float*)d_in[7];
  const float* A_log    = (const float*)d_in[8];
  const float* Dp       = (const float*)d_in[9];
  const float* out_proj = (const float*)d_in[10];
  const float* w1 = (const float*)d_in[11];
  const float* b1 = (const float*)d_in[12];
  const float* w2 = (const float*)d_in[13];
  const float* b2 = (const float*)d_in[14];
  const float* g2 = (const float*)d_in[15];
  const float* bb2 = (const float*)d_in[16];
  const float* w3 = (const float*)d_in[17];
  const float* b3 = (const float*)d_in[18];
  const float* g3 = (const float*)d_in[19];
  const float* bb3 = (const float*)d_in[20];
  const float* w4 = (const float*)d_in[21];
  const float* b4 = (const float*)d_in[22];

  char* ws = (char*)d_ws;
  size_t off = 0;
  auto take = [&](size_t bytes) -> char* {
    char* p = ws + off;
    off += (bytes + 255) & ~(size_t)255;
    return p;
  };

  // Persistent bf16 weights (N,K layout) + mamba output
  bf16* wInProjB  = (bf16*)take((size_t)4096 * 1024 * 2);
  bf16* wXProjB   = (bf16*)take((size_t)96 * 2048 * 2);
  bf16* wDtProjB  = (bf16*)take((size_t)2048 * 64 * 2);
  bf16* wOutProjB = (bf16*)take((size_t)1024 * 2048 * 2);
  bf16* w1t = (bf16*)take((size_t)kH * 512 * 1024 * 2);
  bf16* w2t = (bf16*)take((size_t)kH * 1024 * 512 * 2);
  bf16* w3t = (bf16*)take((size_t)kH * 2048 * 1024 * 2);
  bf16* w4t = (bf16*)take((size_t)kH * kOutPad * 2048 * 2);
  bf16* moB = (bf16*)take((size_t)kM * kDModel * 2);

  const size_t phase = off;
  // Phase A: mamba
  bf16*  uB    = (bf16*) take((size_t)kM * kDModel * 2);
  float* xz    = (float*)take((size_t)kM * 4096 * 4);
  float* xs    = (float*)take((size_t)kM * kDInner * 4);
  bf16*  xsB   = (bf16*) take((size_t)kM * kDInner * 2);
  float* dbl   = (float*)take((size_t)kM * 96 * 4);
  bf16*  dtrB  = (bf16*) take((size_t)kM * 64 * 2);
  float* dtBuf = (float*)take((size_t)kM * kDInner * 4);
  float* yBuf  = (float*)take((size_t)kM * kDInner * 4);
  bf16*  y2B   = (bf16*) take((size_t)kM * kDInner * 2);
  // Phase B: MLP (aliases phase A; safe after out_proj GEMM)
  off = phase;
  bf16*  h1B  = (bf16*) take((size_t)kH * kM * 512 * 2);
  float* h2p  = (float*)take((size_t)kH * kM * 1024 * 4);
  bf16*  h2B  = (bf16*) take((size_t)kH * kM * 1024 * 2);
  float* h3p  = (float*)take((size_t)kH * kM * 2048 * 4);
  bf16*  h3B  = (bf16*) take((size_t)kH * kM * 2048 * 2);

  const dim3 blk(256);
  auto gemmGrid = [](int N, int H) {
    const int nt = (N + 15) / 16;
    return dim3((unsigned)((nt + 3) / 4), (unsigned)(kM / 128), (unsigned)H);
  };
  auto elems = [](long long n) { return dim3((unsigned)((n + 255) / 256)); };

  // --- weight prep ---
  cvt_bf16_kernel<<<elems(4096LL * 1024), blk, 0, stream>>>(in_proj, wInProjB, 4096LL * 1024);
  cvt_bf16_kernel<<<elems(96LL * 2048), blk, 0, stream>>>(x_proj, wXProjB, 96LL * 2048);
  cvt_bf16_kernel<<<elems(2048LL * 64), blk, 0, stream>>>(dt_proj, wDtProjB, 2048LL * 64);
  cvt_bf16_kernel<<<elems(1024LL * 2048), blk, 0, stream>>>(out_proj, wOutProjB, 1024LL * 2048);
  {
    dim3 g1((unsigned)((512LL * 1024 + 255) / 256), 1, kH);
    transpose_w_kernel<<<g1, blk, 0, stream>>>(w1, w1t, 1024, 512, 512);
    dim3 g2d((unsigned)((1024LL * 512 + 255) / 256), 1, kH);
    transpose_w_kernel<<<g2d, blk, 0, stream>>>(w2, w2t, 512, 1024, 1024);
    dim3 g3d((unsigned)((2048LL * 1024 + 255) / 256), 1, kH);
    transpose_w_kernel<<<g3d, blk, 0, stream>>>(w3, w3t, 1024, 2048, 2048);
    dim3 g4d((unsigned)(((long long)kOutPad * 2048 + 255) / 256), 1, kH);
    transpose_w_kernel<<<g4d, blk, 0, stream>>>(w4, w4t, 2048, kOut, kOutPad);
  }

  // --- embedding ---
  embed_kernel<<<dim3(kM), blk, 0, stream>>>(x_ids, emb, uB);

  // --- xz = u @ in_proj.T ---
  gemm_kernel<MODE_F32><<<gemmGrid(4096, 1), blk, 0, stream>>>(
      uB, kDModel, 0, wInProjB, kDModel, 0, nullptr, 0,
      xz, 4096, 0, kM, 4096, kDModel, 4096);

  // --- depthwise conv + silu ---
  conv_silu_kernel<<<elems((long long)kM * kDInner), blk, 0, stream>>>(xz, conv_w, conv_b, xs, xsB);

  // --- dbl = xs @ x_proj.T (N=96) ---
  gemm_kernel<MODE_F32><<<gemmGrid(96, 1), blk, 0, stream>>>(
      xsB, kDInner, 0, wXProjB, kDInner, 0, nullptr, 0,
      dbl, 96, 0, kM, 96, kDInner, 96);

  // --- dt_raw -> bf16 (strided out of dbl) ---
  cvt_strided_kernel<<<elems((long long)kM * 64), blk, 0, stream>>>(dbl, 96, dtrB, 64, kM, 64);

  // --- dt = softplus(dt_raw @ dt_proj.T + b) ---
  gemm_kernel<MODE_SOFTPLUS><<<gemmGrid(kDInner, 1), blk, 0, stream>>>(
      dtrB, 64, 0, wDtProjB, 64, 0, dt_b, 0,
      dtBuf, kDInner, 0, kM, kDInner, 64, kDInner);

  // --- selective scan ---
  scan_kernel<<<dim3(256), blk, 0, stream>>>(dtBuf, xs, dbl, A_log, yBuf);

  // --- gate: (y + xs*D) * silu(z) ---
  ygate_kernel<<<elems((long long)kM * kDInner), blk, 0, stream>>>(yBuf, xs, Dp, xz, y2B);

  // --- mamba out = y2 @ out_proj.T -> bf16 ---
  gemm_kernel<MODE_BF16><<<gemmGrid(kDModel, 1), blk, 0, stream>>>(
      y2B, kDInner, 0, wOutProjB, kDInner, 0, nullptr, 0,
      moB, kDModel, 0, kM, kDModel, kDInner, kDModel);

  // --- head MLPs (grid.z = 4 heads) ---
  gemm_kernel<MODE_BIAS_LEAKY_BF16><<<gemmGrid(512, kH), blk, 0, stream>>>(
      moB, kDModel, 0, w1t, kDModel, (long long)512 * 1024, b1, 512,
      h1B, 512, (long long)kM * 512, kM, 512, kDModel, 512);

  gemm_kernel<MODE_BIAS_LEAKY_F32><<<gemmGrid(1024, kH), blk, 0, stream>>>(
      h1B, 512, (long long)kM * 512, w2t, 512, (long long)1024 * 512, b2, 1024,
      h2p, 1024, (long long)kM * 1024, kM, 1024, 512, 1024);

  layernorm_kernel<<<dim3(kM, kH), blk, 0, stream>>>(
      h2p, (long long)kM * 1024, g2, bb2, 1024, h2B, (long long)kM * 1024, 1024);

  gemm_kernel<MODE_BIAS_LEAKY_F32><<<gemmGrid(2048, kH), blk, 0, stream>>>(
      h2B, 1024, (long long)kM * 1024, w3t, 1024, (long long)2048 * 1024, b3, 2048,
      h3p, 2048, (long long)kM * 2048, kM, 2048, 1024, 2048);

  layernorm_kernel<<<dim3(kM, kH), blk, 0, stream>>>(
      h3p, (long long)kM * 2048, g3, bb3, 2048, h3B, (long long)kM * 2048, 2048);

  // --- logits + fused output transpose into d_out (B,1026,L,H) ---
  gemm_kernel<MODE_LOGITS><<<gemmGrid(kOutPad, kH), blk, 0, stream>>>(
      h3B, 2048, (long long)kM * 2048, w4t, 2048, (long long)kOutPad * 2048, b4, kOut,
      d_out, 0, 0, kM, kOutPad, 2048, kOut);
}
